// OmniGenesisAGI_23587960389766
// MI455X (gfx1250) — matrix operand order repaired and checked by reading, not compiled, not run on hardware
//
#include <hip/hip_runtime.h>
#include <hip/hip_bf16.h>
#include <math.h>
#include <stdint.h>

// ---------------- problem constants (match reference) ----------------
#define VOCAB  50257
#define DIM    1024
#define NEXP   8
#define HID    4096
#define BATCH  4
#define SEQ    512
#define NTOK   (BATCH*SEQ)   // 2048
#define SKETCH 64
#define NBUF   1024

typedef __attribute__((ext_vector_type(16))) __bf16 v16bf;
typedef __attribute__((ext_vector_type(8)))  __bf16 bf16x8;
typedef __attribute__((ext_vector_type(8)))  float  v8f;
typedef __attribute__((ext_vector_type(4)))  unsigned int ui32x4;
typedef __attribute__((ext_vector_type(8)))  int i32x8;
typedef __attribute__((ext_vector_type(4)))  int i32x4;

// LDS B tile: 64 rows x 64 bf16 (one K=64 pipeline stage). TDM inserts a
// 16B pad after every 64B of a row -> row pitch 160B (80 elems); the two
// K=32 chunks start at +0B and +80B, both 16B aligned for ds_load_b128.
#define BROW   80               // LDS row pitch in elements
#define BCHK   40               // element offset of second K-chunk
#define BSTAGE (64 * BROW)      // elements per stage (10KB)

// ---------------- WMMA fragment helpers (wave32, CDNA5 layouts) ------
// A: 16x32 bf16, row-major src (lda in elements).
// lane<16:  M=lane,    K = {0..7, 16..23}
// lane>=16: M=lane-16, K = {8..15, 24..31}
__device__ __forceinline__ v16bf load_a16x32(const __bf16* a, int lda, int lane) {
  int m = lane & 15, h = lane >> 4;
  const __bf16* p = a + m * lda + h * 8;
  bf16x8 lo = *(const bf16x8*)p;
  bf16x8 hi = *(const bf16x8*)(p + 16);
  v16bf r;
#pragma unroll
  for (int i = 0; i < 8; ++i) { r[i] = lo[i]; r[i + 8] = hi[i]; }
  return r;
}
// B fragment from LDS-staged tile (row n holds K contiguous).
// lane<16: N=lane, K={0..7,16..23} ; lane>=16: N=lane-16, K={8..15,24..31}
__device__ __forceinline__ v16bf load_b_lds(const __bf16* bs, int lane) {
  int n = lane & 15, kh = lane >> 4;
  const __bf16* p = bs + n * BROW + kh * 8;
  bf16x8 lo = *(const bf16x8*)p;
  bf16x8 hi = *(const bf16x8*)(p + 16);
  v16bf r;
#pragma unroll
  for (int i = 0; i < 8; ++i) { r[i] = lo[i]; r[i + 8] = hi[i]; }
  return r;
}

// ---------------- Tensor Data Mover: async B-tile -> LDS -------------
// Loads a 64-row x 64-elem bf16 tile from Bt row-major (ldb elems/row)
// into LDS at lds_byte_addr; 16B LDS pad after every 64B (BROW pitch).
// Rows >= nvalid read as zero via TDM OOB semantics (tensor_dim1 = nvalid).
__device__ __forceinline__ void tdm_load_b_tile(const __bf16* gptr, int ldb,
                                                int nvalid,
                                                unsigned lds_byte_addr) {
  unsigned long long ga = (unsigned long long)(uintptr_t)gptr;
  // D# group0: count=1 | lds_addr | global_addr | type=2
  ui32x4 g0 = { 1u,
                lds_byte_addr,
                (unsigned)(ga & 0xFFFFFFFFull),
                (unsigned)((ga >> 32) & 0x01FFFFFFull) | 0x80000000u };
  // D# group1: data_size=2B, pad_enable, pad_interval=16dw(64B), pad_amount=4dw(16B)
  unsigned td0 = 64u, td1 = (unsigned)nvalid, tl0 = 64u, tl1 = 64u;
  unsigned long long s0 = (unsigned long long)(unsigned)ldb;
  i32x8 g1 = { (int)(0x00010000u | (1u << 20) | (3u << 22) | (3u << 25)),
               (int)((td0 & 0xFFFFu) << 16),
               (int)((td0 >> 16) | ((td1 & 0xFFFFu) << 16)),
               (int)((td1 >> 16) | (tl0 << 16)),
               (int)tl1,
               (int)(s0 & 0xFFFFFFFFull),
               (int)((s0 >> 32) & 0xFFFFull),
               0 };
  i32x4 gz = { 0, 0, 0, 0 };
#if defined(__clang_major__) && (__clang_major__ >= 23)
  i32x8 gz8 = { 0, 0, 0, 0, 0, 0, 0, 0 };
  __builtin_amdgcn_tensor_load_to_lds(g0, g1, gz, gz, gz8, 0);
#else
  __builtin_amdgcn_tensor_load_to_lds(g0, g1, gz, gz, 0);
#endif
}

// ---------------- shared WMMA mainloop -------------------------------
// Block = 256 threads (8 waves); wave tile 16(M)x64(N); block 128Mx64N.
// K pipelined in 64-deep stages: B tile double-buffered in LDS via TDM,
// A fragments register-pipelined one stage ahead from global.
__device__ __forceinline__ void wmma_mainloop(const __bf16* __restrict__ A,
                                              int lda, int K,
                                              const __bf16* __restrict__ Bt,
                                              int ldb, int n0, int nvalid,
                                              __bf16* bs, int lane, int wid,
                                              v8f acc[4]) {
  const __bf16* btile = Bt + (size_t)n0 * ldb;
  unsigned lds0 = (unsigned)(uintptr_t)bs;
  if (wid == 0) tdm_load_b_tile(btile, ldb, nvalid, lds0);
  v16bf a0 = load_a16x32(A, lda, lane);
  v16bf a1 = load_a16x32(A + 32, lda, lane);
  v16bf a0n = a0, a1n = a1;
  int cur = 0;
  for (int k0 = 0; k0 < K; k0 += 64) {
    if (wid == 0) {
      if (k0 + 64 < K) {
        tdm_load_b_tile(btile + (k0 + 64), ldb, nvalid,
                        lds0 + (unsigned)((1 - cur) * BSTAGE * 2));
        __builtin_amdgcn_s_wait_tensorcnt(1);   // older DMA (buf[cur]) done
      } else {
        __builtin_amdgcn_s_wait_tensorcnt(0);   // last DMA done
      }
    }
    __syncthreads();                       // buf[cur] ready for all waves
    if (k0 + 64 < K) {
      a0n = load_a16x32(A + k0 + 64, lda, lane);
      a1n = load_a16x32(A + k0 + 96, lda, lane);
    }
    const __bf16* bsc = bs + cur * BSTAGE;
    // K-chunk 0: batch the 4 fragment loads, then 4 WMMAs
    {
      v16bf b0 = load_b_lds(bsc + 0 * 16 * BROW, lane);
      v16bf b1 = load_b_lds(bsc + 1 * 16 * BROW, lane);
      v16bf b2 = load_b_lds(bsc + 2 * 16 * BROW, lane);
      v16bf b3 = load_b_lds(bsc + 3 * 16 * BROW, lane);
      acc[0] = __builtin_amdgcn_wmma_f32_16x16x32_bf16(false, a0, false, b0,
                                                       (short)0, acc[0], false, false);
      acc[1] = __builtin_amdgcn_wmma_f32_16x16x32_bf16(false, a0, false, b1,
                                                       (short)0, acc[1], false, false);
      acc[2] = __builtin_amdgcn_wmma_f32_16x16x32_bf16(false, a0, false, b2,
                                                       (short)0, acc[2], false, false);
      acc[3] = __builtin_amdgcn_wmma_f32_16x16x32_bf16(false, a0, false, b3,
                                                       (short)0, acc[3], false, false);
    }
    // K-chunk 1 (elements 32..63 live at +BCHK in each LDS row)
    {
      v16bf b0 = load_b_lds(bsc + 0 * 16 * BROW + BCHK, lane);
      v16bf b1 = load_b_lds(bsc + 1 * 16 * BROW + BCHK, lane);
      v16bf b2 = load_b_lds(bsc + 2 * 16 * BROW + BCHK, lane);
      v16bf b3 = load_b_lds(bsc + 3 * 16 * BROW + BCHK, lane);
      acc[0] = __builtin_amdgcn_wmma_f32_16x16x32_bf16(false, a1, false, b0,
                                                       (short)0, acc[0], false, false);
      acc[1] = __builtin_amdgcn_wmma_f32_16x16x32_bf16(false, a1, false, b1,
                                                       (short)0, acc[1], false, false);
      acc[2] = __builtin_amdgcn_wmma_f32_16x16x32_bf16(false, a1, false, b2,
                                                       (short)0, acc[2], false, false);
      acc[3] = __builtin_amdgcn_wmma_f32_16x16x32_bf16(false, a1, false, b3,
                                                       (short)0, acc[3], false, false);
    }
    __syncthreads();                       // all reads of buf[cur] done
    a0 = a0n;
    a1 = a1n;
    cur ^= 1;
  }
}

__device__ __forceinline__ float gelu_tanh(float x) {
  float x3 = x * x * x;
  return 0.5f * x * (1.f + tanhf(0.7978845608028654f * (x + 0.044715f * x3)));
}

// ---------------- weight prep ----------------------------------------
__global__ void cvt_bf16_kernel(const float* __restrict__ src,
                                __bf16* __restrict__ dst, int n) {
  int i = blockIdx.x * 256 + threadIdx.x;
  if (i < n) dst[i] = (__bf16)src[i];
}

// src [z][R][C] f32 -> dst [z][C][R] bf16 (32x32 LDS tiles)
__global__ void transpose_cvt_kernel(const float* __restrict__ src,
                                     __bf16* __restrict__ dst, int R, int C) {
  __shared__ float t[32][33];
  const float* s = src + (size_t)blockIdx.z * R * C;
  __bf16* d = dst + (size_t)blockIdx.z * R * C;
  int c0 = blockIdx.x * 32, r0 = blockIdx.y * 32;
#pragma unroll
  for (int i = 0; i < 4; ++i)
    t[threadIdx.y + i * 8][threadIdx.x] =
        s[(size_t)(r0 + threadIdx.y + i * 8) * C + c0 + threadIdx.x];
  __syncthreads();
#pragma unroll
  for (int i = 0; i < 4; ++i)
    d[(size_t)(c0 + threadIdx.y + i * 8) * R + r0 + threadIdx.x] =
        (__bf16)t[threadIdx.x][threadIdx.y + i * 8];
}

// ---------------- embedding gather + bf16 copy -----------------------
__global__ void embed_gather_kernel(const int* __restrict__ ids,
                                    const float* __restrict__ ew,
                                    float* __restrict__ x,
                                    __bf16* __restrict__ xb) {
  int n = blockIdx.x;
  const float* src = ew + (size_t)ids[n] * DIM;
#pragma unroll
  for (int i = 0; i < 4; ++i) {
    int d = threadIdx.x + i * 256;
    float v = src[d];
    x[n * DIM + d] = v;
    xb[n * DIM + d] = (__bf16)v;
  }
}

// mean over sequence: src [B,S,D] -> dst [B,D]
__global__ void mean_s_kernel(const float* __restrict__ src, float* __restrict__ dst) {
  int i = blockIdx.x * 256 + threadIdx.x;
  if (i >= BATCH * DIM) return;
  int b = i >> 10, d = i & 1023;
  float acc = 0.f;
  for (int s = 0; s < SEQ; ++s) acc += src[((b * SEQ + s) << 10) + d];
  dst[i] = acc * (1.f / SEQ);
}

// ---------------- novelty gate ---------------------------------------
__global__ void novelty_kernel(const float* __restrict__ zmean,
                               const float* __restrict__ sw,
                               const float* __restrict__ buf,
                               int* __restrict__ mask) {
  __shared__ float sk[SKETCH];
  __shared__ float red[256];
  __shared__ float snorm;
  int b = blockIdx.x, tid = threadIdx.x;
  if (tid < SKETCH) {
    float a = 0.f;
    for (int d = 0; d < DIM; ++d) a += zmean[b * DIM + d] * sw[d * SKETCH + tid];
    sk[tid] = a;
  }
  __syncthreads();
  red[tid] = (tid < SKETCH) ? sk[tid] * sk[tid] : 0.f;
  __syncthreads();
  for (int s = 128; s > 0; s >>= 1) {
    if (tid < s) red[tid] += red[tid + s];
    __syncthreads();
  }
  if (tid == 0) snorm = 1.f / (sqrtf(red[0]) + 1e-8f);
  __syncthreads();
  float mx = -1e30f;
  for (int i = tid; i < NBUF; i += 256) {
    float dp = 0.f, b2 = 0.f;
    for (int j = 0; j < SKETCH; ++j) {
      float bv = buf[i * SKETCH + j];
      dp += sk[j] * bv;
      b2 += bv * bv;
    }
    mx = fmaxf(mx, dp * snorm / (sqrtf(b2) + 1e-8f));
  }
  red[tid] = mx;
  __syncthreads();
  for (int s = 128; s > 0; s >>= 1) {
    if (tid < s) red[tid] = fmaxf(red[tid], red[tid + s]);
    __syncthreads();
  }
  if (tid == 0) mask[b] = (1.f - red[0]) > 0.5f ? 1 : 0;
}

// ---------------- router: softmax + top-2 + aux stats ----------------
// stats: [0]=sum lse^2, [1..8]=sum probs (imp), [9..16]=routed counts (frac)
__global__ void gate_kernel(const float* __restrict__ x, const float* __restrict__ gw,
                            float* __restrict__ gates, float* __restrict__ stats) {
  __shared__ float red[NEXP][128];
  int n = blockIdx.x, tid = threadIdx.x;
  float s[NEXP];
#pragma unroll
  for (int e = 0; e < NEXP; ++e) s[e] = 0.f;
  for (int d = tid; d < DIM; d += 128) {
    float xv = x[n * DIM + d];
    const float* g = gw + d * NEXP;
#pragma unroll
    for (int e = 0; e < NEXP; ++e) s[e] += xv * g[e];
  }
#pragma unroll
  for (int e = 0; e < NEXP; ++e) red[e][tid] = s[e];
  __syncthreads();
  for (int st = 64; st > 0; st >>= 1) {
    if (tid < st)
#pragma unroll
      for (int e = 0; e < NEXP; ++e) red[e][tid] += red[e][tid + st];
    __syncthreads();
  }
  if (tid == 0) {
    float l[NEXP], mx = -1e30f;
#pragma unroll
    for (int e = 0; e < NEXP; ++e) { l[e] = red[e][0]; mx = fmaxf(mx, l[e]); }
    float se = 0.f, p[NEXP];
#pragma unroll
    for (int e = 0; e < NEXP; ++e) { p[e] = expf(l[e] - mx); se += p[e]; }
    float lse = mx + logf(se);
    atomicAdd(&stats[0], lse * lse);
    float inv = 1.f / se;
#pragma unroll
    for (int e = 0; e < NEXP; ++e) { p[e] *= inv; atomicAdd(&stats[1 + e], p[e]); }
    int i1 = 0;
#pragma unroll
    for (int e = 1; e < NEXP; ++e) if (p[e] > p[i1]) i1 = e;
    int i2 = (i1 == 0) ? 1 : 0;
#pragma unroll
    for (int e = 0; e < NEXP; ++e) if (e != i1 && p[e] > p[i2]) i2 = e;
    float gs = p[i1] + p[i2] + 1e-9f;
#pragma unroll
    for (int e = 0; e < NEXP; ++e) gates[n * NEXP + e] = 0.f;
    gates[n * NEXP + i1] = p[i1] / gs;
    gates[n * NEXP + i2] = p[i2] / gs;
    atomicAdd(&stats[9 + i1], 1.f);
    atomicAdd(&stats[9 + i2], 1.f);
  }
}

// ---------------- MoE GEMMs (bf16 WMMA + TDM-staged B) ---------------
__global__ void gemm_h_kernel(const __bf16* __restrict__ A,   // [NTOK][DIM]
                              const __bf16* __restrict__ Bt,  // w1t[e]: [HID][DIM]
                              const float* __restrict__ bias, // [HID]
                              __bf16* __restrict__ H) {       // [NTOK][HID]
  __shared__ __bf16 Bs[2 * BSTAGE];
  int wid = threadIdx.x >> 5, lane = threadIdx.x & 31;
  int m0 = blockIdx.y * 128 + wid * 16;
  int n0 = blockIdx.x * 64;
  v8f acc[4];
#pragma unroll
  for (int t = 0; t < 4; ++t)
#pragma unroll
    for (int i = 0; i < 8; ++i) acc[t][i] = 0.f;
  wmma_mainloop(A + m0 * DIM, DIM, DIM, Bt, DIM, n0, 64, Bs, lane, wid, acc);
  int col = lane & 15, rb = (lane >> 4) * 8;
#pragma unroll
  for (int t = 0; t < 4; ++t) {
    int n = n0 + t * 16 + col;
    float bn = bias[n];
#pragma unroll
    for (int r = 0; r < 8; ++r)
      H[(m0 + rb + r) * HID + n] = (__bf16)gelu_tanh(acc[t][r] + bn);
  }
}

// xmoe += gates[:,e] * (hbuf @ w2[e] + b2[e])
__global__ void gemm_y_kernel(const __bf16* __restrict__ A,   // hbuf [NTOK][HID]
                              const __bf16* __restrict__ Bt,  // w2t[e]: [DIM][HID]
                              const float* __restrict__ bias, // [DIM]
                              const float* __restrict__ gates,
                              float* __restrict__ xmoe, int e) {
  __shared__ __bf16 Bs[2 * BSTAGE];
  int wid = threadIdx.x >> 5, lane = threadIdx.x & 31;
  int m0 = blockIdx.y * 128 + wid * 16;
  int n0 = blockIdx.x * 64;
  v8f acc[4];
#pragma unroll
  for (int t = 0; t < 4; ++t)
#pragma unroll
    for (int i = 0; i < 8; ++i) acc[t][i] = 0.f;
  wmma_mainloop(A + m0 * HID, HID, HID, Bt, HID, n0, 64, Bs, lane, wid, acc);
  int col = lane & 15, rb = (lane >> 4) * 8;
#pragma unroll
  for (int t = 0; t < 4; ++t) {
    int n = n0 + t * 16 + col;
    float bn = bias[n];
#pragma unroll
    for (int r = 0; r < 8; ++r) {
      int m = m0 + rb + r;
      xmoe[m * DIM + n] += gates[m * NEXP + e] * (acc[t][r] + bn);
    }
  }
}

// logits = final_bf [2048,1024] @ embed_bf^T (embed_bf [VOCAB][DIM] is Bt form)
__global__ void gemm_head_kernel(const __bf16* __restrict__ A,
                                 const __bf16* __restrict__ Bt,
                                 float* __restrict__ out) {
  __shared__ __bf16 Bs[2 * BSTAGE];
  int wid = threadIdx.x >> 5, lane = threadIdx.x & 31;
  int m0 = blockIdx.y * 128 + wid * 16;
  int n0 = blockIdx.x * 64;
  int nv = VOCAB - n0;
  nv = nv > 64 ? 64 : nv;                  // TDM OOB rows -> zeros
  v8f acc[4];
#pragma unroll
  for (int t = 0; t < 4; ++t)
#pragma unroll
    for (int i = 0; i < 8; ++i) acc[t][i] = 0.f;
  wmma_mainloop(A + m0 * DIM, DIM, DIM, Bt, DIM, n0, nv, Bs, lane, wid, acc);
  int col = lane & 15, rb = (lane >> 4) * 8;
#pragma unroll
  for (int t = 0; t < 4; ++t) {
    int n = n0 + t * 16 + col;
    if (n < VOCAB)
#pragma unroll
      for (int r = 0; r < 8; ++r)
        out[(size_t)(m0 + rb + r) * VOCAB + n] = acc[t][r];
  }
}

// ---------------- small dense ops ------------------------------------
__global__ void proj_kernel(const float* __restrict__ mm, const float* __restrict__ pw,
                            const float* __restrict__ pb, float* __restrict__ out) {
  int i = blockIdx.x * 256 + threadIdx.x;
  if (i >= BATCH * DIM) return;
  int b = i >> 10, d = i & 1023;
  float acc = pb[d];
  for (int j = 0; j < DIM; ++j) acc += mm[b * DIM + j] * pw[j * DIM + d];
  out[i] = acc;
}

__global__ void ln_rows_kernel(const float* __restrict__ x, const float* __restrict__ g,
                               const float* __restrict__ bb, float* __restrict__ out) {
  __shared__ float s1[256], s2[256];
  int row = blockIdx.x, tid = threadIdx.x;
  float v[4], ps = 0.f, pq = 0.f;
#pragma unroll
  for (int i = 0; i < 4; ++i) {
    v[i] = x[row * DIM + tid + i * 256];
    ps += v[i];
    pq += v[i] * v[i];
  }
  s1[tid] = ps; s2[tid] = pq;
  __syncthreads();
  for (int s = 128; s > 0; s >>= 1) {
    if (tid < s) { s1[tid] += s1[tid + s]; s2[tid] += s2[tid + s]; }
    __syncthreads();
  }
  float mean = s1[0] * (1.f / DIM);
  float var = s2[0] * (1.f / DIM) - mean * mean;
  float rst = rsqrtf(var + 1e-5f);
#pragma unroll
  for (int i = 0; i < 4; ++i) {
    int d = tid + i * 256;
    out[row * DIM + d] = (v[i] - mean) * rst * g[d] + bb[d];
  }
}

__global__ void reason_step_kernel(const float* __restrict__ zi,
                                   const float* __restrict__ ctx,
                                   const float* __restrict__ rw,
                                   const float* __restrict__ rb,
                                   float* __restrict__ zo) {
  int i = blockIdx.x * 256 + threadIdx.x;
  if (i >= BATCH * DIM) return;
  int b = i >> 10, d = i & 1023;
  float acc = rb[d];
  for (int j = 0; j < DIM; ++j) {
    acc += zi[b * DIM + j] * rw[j * DIM + d];
    acc += ctx[b * DIM + j] * rw[(DIM + j) * DIM + d];
  }
  zo[i] = zi[i] + tanhf(acc);
}

__global__ void conf_kernel(const float* __restrict__ z, const float* __restrict__ cw,
                            const float* __restrict__ cb, float* __restrict__ conf) {
  __shared__ float red[256];
  int b = blockIdx.x, tid = threadIdx.x;
  float p = 0.f;
#pragma unroll
  for (int i = 0; i < 4; ++i) {
    int d = tid + i * 256;
    p += z[b * DIM + d] * cw[d];
  }
  red[tid] = p;
  __syncthreads();
  for (int s = 128; s > 0; s >>= 1) {
    if (tid < s) red[tid] += red[tid + s];
    __syncthreads();
  }
  if (tid == 0) conf[b] = 1.f / (1.f + expf(-(red[0] + cb[0])));
}

__global__ void final_merge_kernel(const float* __restrict__ x,
                                   const float* __restrict__ xmoe,
                                   const float* __restrict__ zfin,
                                   const int* __restrict__ mask,
                                   const float* __restrict__ g,
                                   const float* __restrict__ bb,
                                   __bf16* __restrict__ out) {
  __shared__ float s1[256], s2[256];
  int n = blockIdx.x, tid = threadIdx.x;
  int b = n >> 9;  // SEQ == 512
  int deep = mask[b];
  float v[4], ps = 0.f, pq = 0.f;
#pragma unroll
  for (int i = 0; i < 4; ++i) {
    int d = tid + i * 256;
    float val = deep ? (xmoe[n * DIM + d] + zfin[b * DIM + d]) : x[n * DIM + d];
    v[i] = val;
    ps += val;
    pq += val * val;
  }
  s1[tid] = ps; s2[tid] = pq;
  __syncthreads();
  for (int s = 128; s > 0; s >>= 1) {
    if (tid < s) { s1[tid] += s1[tid + s]; s2[tid] += s2[tid + s]; }
    __syncthreads();
  }
  float mean = s1[0] * (1.f / DIM);
  float var = s2[0] * (1.f / DIM) - mean * mean;
  float rst = rsqrtf(var + 1e-5f);
#pragma unroll
  for (int i = 0; i < 4; ++i) {
    int d = tid + i * 256;
    out[n * DIM + d] = (__bf16)((v[i] - mean) * rst * g[d] + bb[d]);
  }
}

__global__ void finalize_kernel(const float* __restrict__ stats,
                                const float* __restrict__ conf,
                                const int* __restrict__ mask,
                                float* __restrict__ tail) {
  int tid = threadIdx.x;
  if (tid == 0) {
    tail[0] = stats[0] * (1.f / NTOK);             // z_loss
    float aux = 0.f;
    for (int e = 0; e < NEXP; ++e)
      aux += (stats[9 + e] * (1.f / NTOK)) * (stats[1 + e] * (1.f / NTOK));
    tail[1] = (float)NEXP * aux;                   // aux_loss
  }
  if (tid < BATCH) tail[2 + tid] = mask[tid] ? conf[tid] : 0.f;
}

// ---------------- host orchestration ---------------------------------
extern "C" void kernel_launch(void* const* d_in, const int* in_sizes, int n_in,
                              void* d_out, int out_size, void* d_ws, size_t ws_size,
                              hipStream_t stream) {
  (void)in_sizes; (void)n_in; (void)out_size; (void)ws_size;
  const int*   ids     = (const int*)d_in[0];
  const float* embed_w = (const float*)d_in[1];
  const float* norm_g  = (const float*)d_in[2];
  const float* norm_b  = (const float*)d_in[3];
  const float* proj_w  = (const float*)d_in[4];
  const float* proj_b  = (const float*)d_in[5];
  const float* pln_g   = (const float*)d_in[6];
  const float* pln_b   = (const float*)d_in[7];
  const float* gate_w  = (const float*)d_in[8];
  const float* e_w1    = (const float*)d_in[9];
  const float* e_b1    = (const float*)d_in[10];
  const float* e_w2    = (const float*)d_in[11];
  const float* e_b2    = (const float*)d_in[12];
  const float* rw      = (const float*)d_in[13];
  const float* rb      = (const float*)d_in[14];
  const float* cw      = (const float*)d_in[15];
  const float* cb      = (const float*)d_in[16];
  const float* sw      = (const float*)d_in[17];
  const float* nbuf    = (const float*)d_in[18];
  float* out = (float*)d_out;

  char* base = (char*)d_ws;
  size_t o = 0;
  auto carve = [&](size_t bytes) -> char* {
    char* p = base + o;
    o = (o + bytes + 255) & ~(size_t)255;
    return p;
  };
  __bf16* embed_bf = (__bf16*)carve((size_t)VOCAB * DIM * 2);
  __bf16* w1t      = (__bf16*)carve((size_t)NEXP * HID * DIM * 2);
  __bf16* w2t      = (__bf16*)carve((size_t)NEXP * DIM * HID * 2);
  __bf16* xb       = (__bf16*)carve((size_t)NTOK * DIM * 2);
  float*  xf       = (float*) carve((size_t)NTOK * DIM * 4);
  __bf16* hbuf     = (__bf16*)carve((size_t)NTOK * HID * 2);
  float*  xmoe     = (float*) carve((size_t)NTOK * DIM * 4);
  __bf16* finalbf  = (__bf16*)carve((size_t)NTOK * DIM * 2);
  float*  gates    = (float*) carve((size_t)NTOK * NEXP * 4);
  float*  zmean    = (float*) carve((size_t)BATCH * DIM * 4);
  float*  mmoe     = (float*) carve((size_t)BATCH * DIM * 4);
  float*  zpre     = (float*) carve((size_t)BATCH * DIM * 4);
  float*  z0       = (float*) carve((size_t)BATCH * DIM * 4);
  float*  z1       = (float*) carve((size_t)BATCH * DIM * 4);
  float*  conf     = (float*) carve(BATCH * 4);
  float*  stats    = (float*) carve(17 * 4);
  int*    mask     = (int*)   carve(BATCH * 4);

  (void)hipMemsetAsync(stats, 0, 17 * 4, stream);
  (void)hipMemsetAsync(xmoe, 0, (size_t)NTOK * DIM * 4, stream);

  // weight prep: bf16 convert (+transpose so WMMA B operands are K-contiguous)
  {
    int n = VOCAB * DIM;
    cvt_bf16_kernel<<<(n + 255) / 256, 256, 0, stream>>>(embed_w, embed_bf, n);
  }
  transpose_cvt_kernel<<<dim3(HID / 32, DIM / 32, NEXP), dim3(32, 8), 0, stream>>>(
      e_w1, w1t, DIM, HID);   // [E][D][H] -> [E][H][D]
  transpose_cvt_kernel<<<dim3(DIM / 32, HID / 32, NEXP), dim3(32, 8), 0, stream>>>(
      e_w2, w2t, HID, DIM);   // [E][H][D] -> [E][D][H]

  // embed + sequence summary + novelty gate
  embed_gather_kernel<<<NTOK, 256, 0, stream>>>(ids, embed_w, xf, xb);
  mean_s_kernel<<<(BATCH * DIM + 255) / 256, 256, 0, stream>>>(xf, zmean);
  novelty_kernel<<<BATCH, 256, 0, stream>>>(zmean, sw, nbuf, mask);

  // router
  gate_kernel<<<NTOK, 128, 0, stream>>>(xf, gate_w, gates, stats);

  // MoE: per-expert WMMA GEMM pair (TDM-staged B tiles)
  for (int e = 0; e < NEXP; ++e) {
    gemm_h_kernel<<<dim3(HID / 64, NTOK / 128), 256, 0, stream>>>(
        xb, w1t + (size_t)e * HID * DIM, e_b1 + e * HID, hbuf);
    gemm_y_kernel<<<dim3(DIM / 64, NTOK / 128), 256, 0, stream>>>(
        hbuf, w2t + (size_t)e * DIM * HID, e_b2 + e * DIM, gates, xmoe, e);
  }

  // reasoning path
  mean_s_kernel<<<(BATCH * DIM + 255) / 256, 256, 0, stream>>>(xmoe, mmoe);
  proj_kernel<<<(BATCH * DIM + 255) / 256, 256, 0, stream>>>(mmoe, proj_w, proj_b, zpre);
  ln_rows_kernel<<<BATCH, 256, 0, stream>>>(zpre, pln_g, pln_b, z0);
  reason_step_kernel<<<(BATCH * DIM + 255) / 256, 256, 0, stream>>>(z0, zmean, rw, rb, z1);
  reason_step_kernel<<<(BATCH * DIM + 255) / 256, 256, 0, stream>>>(z1, zmean, rw, rb, z0);
  reason_step_kernel<<<(BATCH * DIM + 255) / 256, 256, 0, stream>>>(z0, zmean, rw, rb, z1);
  reason_step_kernel<<<(BATCH * DIM + 255) / 256, 256, 0, stream>>>(z1, zmean, rw, rb, z0);
  conf_kernel<<<BATCH, 256, 0, stream>>>(z0, cw, cb, conf);

  // merge + final LN -> bf16 activations for the tied head
  final_merge_kernel<<<NTOK, 256, 0, stream>>>(xf, xmoe, z0, mask, norm_g, norm_b, finalbf);

  // tied head GEMM (bf16 WMMA) -> logits f32
  gemm_head_kernel<<<dim3((VOCAB + 63) / 64, NTOK / 128), 256, 0, stream>>>(
      finalbf, embed_bf, out);

  // scalar outputs appended after logits
  finalize_kernel<<<1, 32, 0, stream>>>(stats, conf, mask,
                                        out + (size_t)NTOK * VOCAB);
}
